// GroupedQueryAttention_15822659519063
// MI455X (gfx1250) — matrix operand order repaired and checked by reading, not compiled, not run on hardware
//
#include <hip/hip_runtime.h>

// ---------------------------------------------------------------------------
// GQA block for MI455X (gfx1250): bf16 WMMA everywhere, fp32 accumulate.
//   Q = x Wq^T ; K = x Wk^T ; V = x Wv^T            (WMMA GEMM, bf16 out)
//   Vt = transpose(V) per KV head                    (one-time, cheap)
//   attn: flash-style softmax(QK^T/8) V per head     (WMMA + async LDS copy)
//   out = ctx Wo^T                                   (WMMA GEMM, f32 out)
// All LDS fragment traffic is contiguous (ds_load_b128); every GEMM/attention
// tile is staged with global_load_async_to_lds_b128 (ASYNCcnt).
// ---------------------------------------------------------------------------

typedef __bf16 bf16;
typedef __attribute__((ext_vector_type(4)))  __bf16 bf16x4;
typedef __attribute__((ext_vector_type(16))) __bf16 bf16x16;
typedef __attribute__((ext_vector_type(8)))  float  floatx8;

#define BATCH      2
#define SEQ        2048
#define DMODEL     2048
#define NHEADS     32
#define NKV        8
#define DHEAD      64
#define DKV        (NKV * DHEAD)       // 512
#define MSROWS     (BATCH * SEQ)       // 4096
#define SCALE      0.125f              // 1/sqrt(64)

// -------------------------- WMMA helpers -----------------------------------

__device__ __forceinline__ floatx8 wmma_bf16(bf16x16 a, bf16x16 b, floatx8 c) {
  return __builtin_amdgcn_wmma_f32_16x16x32_bf16(
      /*neg_a=*/false, a, /*neg_b=*/false, b,
      /*c_mod=*/(short)0, c, /*reuse_a=*/false, /*reuse_b=*/false);
}

// A fragment, row-major [M][K] in LDS (ISA 7.12.2 16-bit A layout).
__device__ __forceinline__ bf16x16 frag_a_rowmajor(const bf16* s, int ld,
                                                   int row0, int k0, int lane) {
  const int r  = row0 + (lane & 15);
  const int kh = ((lane >> 4) & 1) * 8;
  const bf16* p = s + r * ld + k0 + kh;
  bf16x16 f;
#pragma unroll
  for (int e = 0; e < 8; ++e) f[e] = p[e];
#pragma unroll
  for (int e = 0; e < 8; ++e) f[8 + e] = p[16 + e];
  return f;
}

// B fragment, B[k][n] stored N-major in LDS: s[n][k] (contiguous per lane).
__device__ __forceinline__ bf16x16 frag_b_nmajor(const bf16* s, int ld,
                                                 int col0, int k0, int lane) {
  const int n  = col0 + (lane & 15);
  const int kk = k0 + ((lane >> 4) & 1) * 8;
  const bf16* p = s + n * ld + kk;
  bf16x16 f;
#pragma unroll
  for (int e = 0; e < 8; ++e) f[e] = p[e];
#pragma unroll
  for (int e = 0; e < 8; ++e) f[8 + e] = p[16 + e];
  return f;
}

// -------------------------- CDNA5 async LDS copy ---------------------------
// GLOBAL_LOAD_ASYNC_TO_LDS_B128 (ISA 15.18.3 op 98), tracked by ASYNCcnt.
// Per-lane: copies 16 bytes from global vaddr to LDS address vdst. LDS offset = low 32 bits of the
// flat shared-memory address (aperture rule, ISA 10.2).

__device__ __forceinline__ void async_copy_b128(const bf16* gp, bf16* lp) {
  const unsigned long long ga = (unsigned long long)(uintptr_t)gp;
  const unsigned la = (unsigned)(uintptr_t)lp;
  asm volatile("global_load_async_to_lds_b128 %0, %1, off"
               :: "v"(la), "v"(ga) : "memory");
}

__device__ __forceinline__ void async_wait0() {
  asm volatile("s_wait_asynccnt 0x0" ::: "memory");
}

// -------------------------- fp32 -> bf16 convert (x4 vectorized) -----------

__global__ __launch_bounds__(256) void cvt_f32_bf16_v4(const float4* __restrict__ in,
                                                       bf16x4* __restrict__ out, int n4) {
  int i = blockIdx.x * blockDim.x + threadIdx.x;
  const int stride = gridDim.x * blockDim.x;
  for (; i < n4; i += stride) {
    const float4 v = in[i];
    bf16x4 o = {(bf16)v.x, (bf16)v.y, (bf16)v.z, (bf16)v.w};
    out[i] = o;
  }
}

// -------------------------- V transpose ------------------------------------
// Vt[(b*NKV+g)*64 + d][s] = V[(b*SEQ+s)*DKV + g*64 + d]

__global__ __launch_bounds__(256) void transpose_v(const bf16* __restrict__ V,
                                                   bf16* __restrict__ Vt) {
  constexpr int LD = 72;
  __shared__ __align__(16) bf16 t[64 * LD];
  const int sb  = blockIdx.x * 64;   // seq block
  const int bg  = blockIdx.y;        // b*NKV + g
  const int b   = bg >> 3;
  const int g   = bg & 7;
  const int tid = threadIdx.x;
#pragma unroll
  for (int c = 0; c < 2; ++c) {
    const int ch = tid + c * 256;
    const int r  = ch >> 3;          // seq within tile
    const int cc = (ch & 7) * 8;     // d chunk
    __align__(16) bf16 tmp[8];
    *(int4*)tmp = *(const int4*)(V + ((size_t)(b * SEQ + sb + r)) * DKV + g * DHEAD + cc);
#pragma unroll
    for (int j = 0; j < 8; ++j) t[(cc + j) * LD + r] = tmp[j];
  }
  __syncthreads();
#pragma unroll
  for (int c = 0; c < 2; ++c) {
    const int ch = tid + c * 256;
    const int r  = ch >> 3;          // d row
    const int cc = (ch & 7) * 8;     // seq chunk
    *(int4*)(Vt + ((size_t)(bg * 64 + r)) * SEQ + sb + cc) = *(const int4*)(t + r * LD + cc);
  }
}

// -------------------------- generic WMMA GEMM ------------------------------
// C[M,N] = A[M,K] @ Bt[N,K]^T.  Block tile 64x128x64, 8 waves of 32x32.
// Tiles staged via async LDS copy; next A tile prefetched into GL2.

template <bool OUT_F32>
__global__ __launch_bounds__(256) void gemm_bf16_nt(const bf16* __restrict__ A,
                                                    const bf16* __restrict__ Bt,
                                                    void* __restrict__ Cv,
                                                    int M, int N, int K) {
  constexpr int TM = 64, TN = 128, TK = 64;
  constexpr int LDT = TK + 8;   // 72, keeps 16B alignment, staggers banks
  __shared__ __align__(16) bf16 sA[TM * LDT];
  __shared__ __align__(16) bf16 sB[TN * LDT];

  const int tid  = threadIdx.x;
  const int lane = tid & 31;
  const int wave = tid >> 5;
  const int wm   = (wave & 1) * 32;
  const int wn   = (wave >> 1) * 32;

  const int bm = blockIdx.x * TM;
  const int bn = blockIdx.y * TN;

  const floatx8 zero = {};
  floatx8 acc00 = zero, acc01 = zero, acc10 = zero, acc11 = zero;

  for (int k0 = 0; k0 < K; k0 += TK) {
    // ---- stage A tile 64x64 (async) ----
#pragma unroll
    for (int c = 0; c < 2; ++c) {
      const int ch = tid + c * 256;
      const int r  = ch >> 3;
      const int cc = (ch & 7) * 8;
      const size_t goff = (size_t)(bm + r) * K + k0 + cc;
      async_copy_b128(A + goff, sA + r * LDT + cc);
      if (k0 + TK < K) __builtin_prefetch(A + goff + TK, 0, 1);
    }
    // ---- stage B tile 128x64 (async) ----
#pragma unroll
    for (int c = 0; c < 4; ++c) {
      const int ch = tid + c * 256;
      const int r  = ch >> 3;
      const int cc = (ch & 7) * 8;
      async_copy_b128(Bt + (size_t)(bn + r) * K + k0 + cc, sB + r * LDT + cc);
    }
    async_wait0();
    __syncthreads();

#pragma unroll
    for (int ks = 0; ks < TK; ks += 32) {
      const bf16x16 a0 = frag_a_rowmajor(sA, LDT, wm,      ks, lane);
      const bf16x16 a1 = frag_a_rowmajor(sA, LDT, wm + 16, ks, lane);
      const bf16x16 b0 = frag_b_nmajor(sB, LDT, wn,      ks, lane);
      const bf16x16 b1 = frag_b_nmajor(sB, LDT, wn + 16, ks, lane);
      acc00 = wmma_bf16(a0, b0, acc00);
      acc01 = wmma_bf16(a0, b1, acc01);
      acc10 = wmma_bf16(a1, b0, acc10);
      acc11 = wmma_bf16(a1, b1, acc11);
    }
    __syncthreads();
  }

  const int hsel = (lane >> 4) & 1;
  const int ccol = lane & 15;
#pragma unroll
  for (int r = 0; r < 8; ++r) {
    const int rowl = r + 8 * hsel;
    const size_t row0 = (size_t)(bm + wm + rowl) * N + bn + wn + ccol;
    const size_t row1 = (size_t)(bm + wm + 16 + rowl) * N + bn + wn + ccol;
    if constexpr (OUT_F32) {
      float* C = (float*)Cv;
      C[row0]      = acc00[r];
      C[row0 + 16] = acc01[r];
      C[row1]      = acc10[r];
      C[row1 + 16] = acc11[r];
    } else {
      bf16* C = (bf16*)Cv;
      C[row0]      = (bf16)acc00[r];
      C[row0 + 16] = (bf16)acc01[r];
      C[row1]      = (bf16)acc10[r];
      C[row1 + 16] = (bf16)acc11[r];
    }
  }
}

// -------------------------- GQA flash attention ----------------------------
// One workgroup per (batch, head, 64-query block). 8 waves: mt = wave&3 picks
// the 16-row query tile, nh = wave>>2 picks two 16-wide column tiles.
// Softmax is parallel: 4 threads per row (16 cols each) + LDS reduction.

__global__ __launch_bounds__(256) void gqa_attention(const bf16* __restrict__ Qm,
                                                     const bf16* __restrict__ Km,
                                                     const bf16* __restrict__ Vt,
                                                     bf16* __restrict__ Ctx) {
  constexpr int LD   = 72;
  constexpr int LDSS = 68;
  __shared__ __align__(16) bf16  sQ[64 * LD];
  __shared__ __align__(16) bf16  sK[64 * LD];   // [seq][d]
  __shared__ __align__(16) bf16  sV[64 * LD];   // transposed: [d][seq]
  __shared__ __align__(16) bf16  sP[64 * LD];
  __shared__ __align__(16) float sS[64 * LDSS];
  __shared__ float sM[64], sL[64], sAl[64];
  __shared__ float sRmax[256];
  __shared__ float sRsum[256];

  int idx = blockIdx.x;
  const int qb = idx & 31;  idx >>= 5;   // SEQ/64 query blocks
  const int h  = idx & 31;  idx >>= 5;   // head
  const int b  = idx;                    // batch
  const int g  = h >> 2;                 // KV group
  const int bg = b * NKV + g;

  const int tid  = threadIdx.x;
  const int lane = tid & 31;
  const int wave = tid >> 5;
  const int mt   = wave & 3;
  const int nh   = wave >> 2;

  // ---- stage Q tile (64x64) via async copy ----
#pragma unroll
  for (int c = 0; c < 2; ++c) {
    const int ch = tid + c * 256;
    const int r  = ch >> 3;
    const int cc = (ch & 7) * 8;
    async_copy_b128(Qm + ((size_t)(b * SEQ + qb * 64 + r)) * DMODEL + h * DHEAD + cc,
                    sQ + r * LD + cc);
  }
  if (tid < 64) { sM[tid] = -1e30f; sL[tid] = 0.f; }

  const floatx8 zero = {};
  floatx8 o0 = zero, o1 = zero;
  async_wait0();
  __syncthreads();

  for (int kb = 0; kb < SEQ / 64; ++kb) {
    // ---- stage K [seq][d] and Vt [d][seq] tiles via async copy ----
#pragma unroll
    for (int c = 0; c < 2; ++c) {
      const int ch = tid + c * 256;
      const int r  = ch >> 3;
      const int cc = (ch & 7) * 8;
      async_copy_b128(Km + ((size_t)(b * SEQ + kb * 64 + r)) * DKV + g * DHEAD + cc,
                      sK + r * LD + cc);
      async_copy_b128(Vt + ((size_t)(bg * 64 + r)) * SEQ + kb * 64 + cc,
                      sV + r * LD + cc);
    }
    async_wait0();
    __syncthreads();

    // ---- scores S = Q K^T : tiles (mt, 2nh) and (mt, 2nh+1) ----
    const bf16x16 aq0 = frag_a_rowmajor(sQ, LD, mt * 16, 0,  lane);
    const bf16x16 aq1 = frag_a_rowmajor(sQ, LD, mt * 16, 32, lane);
    floatx8 s0 = zero, s1 = zero;
    {
      bf16x16 bk;
      bk = frag_b_nmajor(sK, LD, (nh * 2) * 16,     0,  lane); s0 = wmma_bf16(aq0, bk, s0);
      bk = frag_b_nmajor(sK, LD, (nh * 2) * 16,     32, lane); s0 = wmma_bf16(aq1, bk, s0);
      bk = frag_b_nmajor(sK, LD, (nh * 2 + 1) * 16, 0,  lane); s1 = wmma_bf16(aq0, bk, s1);
      bk = frag_b_nmajor(sK, LD, (nh * 2 + 1) * 16, 32, lane); s1 = wmma_bf16(aq1, bk, s1);
    }
    {
      const int hsel = (lane >> 4) & 1;
      const int ccol = lane & 15;
#pragma unroll
      for (int r = 0; r < 8; ++r) {
        const int row = mt * 16 + r + 8 * hsel;
        sS[row * LDSS + (nh * 2) * 16 + ccol]     = s0[r] * SCALE;
        sS[row * LDSS + (nh * 2 + 1) * 16 + ccol] = s1[r] * SCALE;
      }
    }
    __syncthreads();

    // ---- online softmax, 4 threads per row ----
    {
      const int row  = tid >> 2;
      const int quad = tid & 3;
      const float* srow = sS + row * LDSS + quad * 16;
      float pm = -1e30f;
#pragma unroll
      for (int j = 0; j < 16; ++j) pm = fmaxf(pm, srow[j]);
      sRmax[tid] = pm;
    }
    __syncthreads();
    {
      const int row  = tid >> 2;
      const int quad = tid & 3;
      const float mold = sM[row];
      const float mx = fmaxf(fmaxf(fmaxf(sRmax[row * 4 + 0], sRmax[row * 4 + 1]),
                                   fmaxf(sRmax[row * 4 + 2], sRmax[row * 4 + 3])),
                             mold);
      const float* srow = sS + row * LDSS + quad * 16;
      bf16* prow = sP + row * LD + quad * 16;
      float psum = 0.f;
#pragma unroll
      for (int j = 0; j < 16; ++j) {
        const float pv = __expf(srow[j] - mx);
        prow[j] = (bf16)pv;
        psum += pv;
      }
      sRsum[tid] = psum;
      __syncthreads();
      if (quad == 0) {
        const float alpha = __expf(mold - mx);
        const float sum = sRsum[row * 4 + 0] + sRsum[row * 4 + 1] +
                          sRsum[row * 4 + 2] + sRsum[row * 4 + 3];
        sM[row]  = mx;
        sL[row]  = sL[row] * alpha + sum;
        sAl[row] = alpha;
      }
    }
    __syncthreads();

    // ---- rescale running O, then O += P @ V (B from transposed sV) ----
    {
      const int hsel = (lane >> 4) & 1;
#pragma unroll
      for (int r = 0; r < 8; ++r) {
        const float alpha = sAl[mt * 16 + r + 8 * hsel];
        o0[r] *= alpha;
        o1[r] *= alpha;
      }
    }
    const bf16x16 ap0 = frag_a_rowmajor(sP, LD, mt * 16, 0,  lane);
    const bf16x16 ap1 = frag_a_rowmajor(sP, LD, mt * 16, 32, lane);
    {
      bf16x16 bv;
      bv = frag_b_nmajor(sV, LD, (nh * 2) * 16,     0,  lane); o0 = wmma_bf16(ap0, bv, o0);
      bv = frag_b_nmajor(sV, LD, (nh * 2) * 16,     32, lane); o0 = wmma_bf16(ap1, bv, o0);
      bv = frag_b_nmajor(sV, LD, (nh * 2 + 1) * 16, 0,  lane); o1 = wmma_bf16(ap0, bv, o1);
      bv = frag_b_nmajor(sV, LD, (nh * 2 + 1) * 16, 32, lane); o1 = wmma_bf16(ap1, bv, o1);
    }
    __syncthreads();  // protect sK/sV/sP before next iteration's staging
  }

  // ---- normalize and write context (head-concatenated layout) ----
  const int hsel = (lane >> 4) & 1;
  const int ccol = lane & 15;
#pragma unroll
  for (int r = 0; r < 8; ++r) {
    const int rloc = mt * 16 + r + 8 * hsel;
    const float inv = 1.0f / sL[rloc];
    const size_t rowg = (size_t)(b * SEQ + qb * 64 + rloc) * DMODEL
                      + (size_t)h * DHEAD + nh * 32 + ccol;
    Ctx[rowg]      = (bf16)(o0[r] * inv);
    Ctx[rowg + 16] = (bf16)(o1[r] * inv);
  }
}

// -------------------------- host launcher ----------------------------------

extern "C" void kernel_launch(void* const* d_in, const int* in_sizes, int n_in,
                              void* d_out, int out_size, void* d_ws, size_t ws_size,
                              hipStream_t stream) {
  (void)in_sizes; (void)n_in; (void)out_size; (void)ws_size;

  const float* x  = (const float*)d_in[0];
  const float* Wq = (const float*)d_in[1];
  const float* Wk = (const float*)d_in[2];
  const float* Wv = (const float*)d_in[3];
  const float* Wo = (const float*)d_in[4];
  float* out = (float*)d_out;

  // workspace carve-up (bf16 copies + intermediates, ~80 MB)
  char* p = (char*)d_ws;
  bf16* xb  = (bf16*)p; p += (size_t)MSROWS * DMODEL * sizeof(bf16);
  bf16* wq  = (bf16*)p; p += (size_t)DMODEL * DMODEL * sizeof(bf16);
  bf16* wk  = (bf16*)p; p += (size_t)DKV    * DMODEL * sizeof(bf16);
  bf16* wv  = (bf16*)p; p += (size_t)DKV    * DMODEL * sizeof(bf16);
  bf16* wo  = (bf16*)p; p += (size_t)DMODEL * DMODEL * sizeof(bf16);
  bf16* Qb  = (bf16*)p; p += (size_t)MSROWS * DMODEL * sizeof(bf16);
  bf16* Kb  = (bf16*)p; p += (size_t)MSROWS * DKV    * sizeof(bf16);
  bf16* Vb  = (bf16*)p; p += (size_t)MSROWS * DKV    * sizeof(bf16);
  bf16* Vtb = (bf16*)p; p += (size_t)MSROWS * DKV    * sizeof(bf16);
  bf16* Cx  = (bf16*)p; p += (size_t)MSROWS * DMODEL * sizeof(bf16);

  auto cvt = [&](const float* src, bf16* dst, size_t n) {
    const int n4 = (int)(n / 4);
    int blocks = (n4 + 255) / 256;
    if (blocks > 8192) blocks = 8192;
    cvt_f32_bf16_v4<<<blocks, 256, 0, stream>>>((const float4*)src, (bf16x4*)dst, n4);
  };
  cvt(x,  xb, (size_t)MSROWS * DMODEL);
  cvt(Wq, wq, (size_t)DMODEL * DMODEL);
  cvt(Wk, wk, (size_t)DKV    * DMODEL);
  cvt(Wv, wv, (size_t)DKV    * DMODEL);
  cvt(Wo, wo, (size_t)DMODEL * DMODEL);

  const dim3 blk(256);
  // Q/K/V projections (y = x @ W^T), bf16 outputs
  gemm_bf16_nt<false><<<dim3(MSROWS / 64, DMODEL / 128), blk, 0, stream>>>(
      xb, wq, Qb, MSROWS, DMODEL, DMODEL);
  gemm_bf16_nt<false><<<dim3(MSROWS / 64, DKV / 128), blk, 0, stream>>>(
      xb, wk, Kb, MSROWS, DKV, DMODEL);
  gemm_bf16_nt<false><<<dim3(MSROWS / 64, DKV / 128), blk, 0, stream>>>(
      xb, wv, Vb, MSROWS, DKV, DMODEL);

  // per-(batch, group) V transpose: Vt[bg*64 + d][s]
  transpose_v<<<dim3(SEQ / 64, BATCH * NKV), blk, 0, stream>>>(Vb, Vtb);

  // attention: 2 batches * 32 heads * 32 query blocks = 2048 workgroups
  gqa_attention<<<dim3(BATCH * NHEADS * (SEQ / 64)), blk, 0, stream>>>(Qb, Kb, Vtb, Cx);

  // output projection, f32 straight into d_out
  gemm_bf16_nt<true><<<dim3(MSROWS / 64, DMODEL / 128), blk, 0, stream>>>(
      Cx, wo, out, MSROWS, DMODEL, DMODEL);
}
